// GlobalAdaptiveDecoder_35012573397125
// MI455X (gfx1250) — matrix operand order repaired and checked by reading, not compiled, not run on hardware
//
#include <hip/hip_runtime.h>
#include <hip/hip_bf16.h>

// ---------------------------------------------------------------------------
// Types & helpers
// ---------------------------------------------------------------------------
typedef __attribute__((ext_vector_type(16))) __bf16 v16bf;
typedef __attribute__((ext_vector_type(8)))  float  v8f;

union Frag {
  v16bf v;
  unsigned int   u[8];
  unsigned short us[16];
};

__device__ __forceinline__ unsigned short f2bf(float f) {
  union { float f; unsigned int u; } c; c.f = f;
  unsigned int u = c.u;
  unsigned int r = (u + 0x7FFFu + ((u >> 16) & 1u)) >> 16;   // RNE
  return (unsigned short)r;
}
__device__ __forceinline__ float bf2f(unsigned short h) {
  union { unsigned int u; float f; } c; c.u = ((unsigned int)h) << 16;
  return c.f;
}
__device__ __forceinline__ v8f wmma_bf16(const Frag& a, const Frag& b, v8f c) {
  return __builtin_amdgcn_wmma_f32_16x16x32_bf16(
      false, a.v, false, b.v, (short)0, c, false, false);
}

// ---- CDNA5 async global->LDS staging (guarded; sync fallback otherwise) ----
#if defined(__gfx1250__) && __has_builtin(__builtin_amdgcn_global_load_async_to_lds_b128)
#define GEMM_ASYNC 1
typedef int v4i __attribute__((vector_size(4 * sizeof(int))));
#if __has_builtin(__builtin_amdgcn_s_wait_asynccnt)
#define WAIT_ASYNC(n) __builtin_amdgcn_s_wait_asynccnt(n)
#else
#define WAIT_ASYNC(n) asm volatile("s_wait_asynccnt %0" ::"i"(n) : "memory")
#endif
#else
#define GEMM_ASYNC 0
#endif

// ---------------------------------------------------------------------------
// fp32 -> bf16 conversion, grid-stride
// ---------------------------------------------------------------------------
__global__ void k_f32_to_bf16(const float* __restrict__ in,
                              unsigned short* __restrict__ out, int n) {
  int i = blockIdx.x * blockDim.x + threadIdx.x;
  int stride = gridDim.x * blockDim.x;
  for (; i < n; i += stride) out[i] = f2bf(in[i]);
}

// ---------------------------------------------------------------------------
// Embedding + sinusoidal positional encoding -> bf16 [B*S, D]
// ---------------------------------------------------------------------------
__global__ void k_embed(const int* __restrict__ x, const float* __restrict__ emb,
                        unsigned short* __restrict__ X, int S, int Dd) {
  int bs = blockIdx.x;
  int s  = bs % S;
  int tok = x[bs];
  float lg = __logf(10000.0f);
  for (int d = threadIdx.x; d < Dd; d += blockDim.x) {
    int j = d >> 1;
    float ang = (float)s * __expf(-((float)(2 * j) / (float)Dd) * lg);
    float pe  = (d & 1) ? __cosf(ang) : __sinf(ang);
    X[(size_t)bs * Dd + d] = f2bf(emb[(size_t)tok * Dd + d] + pe);
  }
}

// ---------------------------------------------------------------------------
// Tiled bf16 WMMA GEMM: C[M,N] = A[M,K]*B[K,N] (+rowbias, +fbias)
// 256 thr (8 waves); 128x64 C tile; each wave computes 32x32 (4 WMMAs/step).
// A: async global->LDS 4-deep ring (2 b128/thread/tile, ASYNCcnt pipelined).
// B: register double-buffer, stored transposed -> dword fragment loads.
// Pipeline peeled: branch-free steady-state loop + 2 tail steps.
// One barrier per k-step (ring depths bound wave skew safely).
// MG/NG: compile-out M/N bounds checks for full-tile instances.
// ---------------------------------------------------------------------------
template <int MG, int NG>
__global__ __launch_bounds__(256) void k_gemm(
    const unsigned short* __restrict__ A, const unsigned short* __restrict__ B,
    void* __restrict__ Cv, int M, int N, int K,
    const unsigned short* __restrict__ rowbias, int rb_shift,
    const float* __restrict__ fbias, int out_f32) {
  __shared__ unsigned short As[4][128][40];  // [ring][row][k] 16B-aligned rows
  __shared__ unsigned short BsT[2][64][34];  // [ring][n][k]  dword frag loads

  int tid = threadIdx.x;
  int lane = tid & 31;
  int w = tid >> 5;
  int wr = w & 3, wc = w >> 2;
  int hi = lane >> 4, l15 = lane & 15;
  int m0 = blockIdx.y * 128;
  int n0 = blockIdx.x * 64;

  int ra0 = tid >> 2, ca = (tid & 3) * 8;    // A vec 0: rows 0..63
  int ra1 = ra0 + 64;                        // A vec 1: rows 64..127
  int r_b = tid >> 3, c_b = (tid & 7) * 8;   // B stage: 32 k x 64 n
  int nt = K >> 5;

  v8f acc[2][2] = {{{}, {}}, {{}, {}}};
  unsigned bu[4];
#if !GEMM_ASYNC
  unsigned au[8];
#endif

  auto loadB = [&](int k0) {
    if (!NG || n0 + 64 <= N) {
      const uint4* p = (const uint4*)(B + (size_t)(k0 + r_b) * N + n0 + c_b);
      uint4 t = *p;
      bu[0] = t.x; bu[1] = t.y; bu[2] = t.z; bu[3] = t.w;
      __builtin_prefetch(B + (size_t)(k0 + 32 + r_b) * N + n0 + c_b, 0, 1);
    } else {
      #pragma unroll
      for (int j = 0; j < 4; ++j) {
        int c0 = n0 + c_b + 2 * j, c1 = c0 + 1;
        unsigned lo  = (c0 < N) ? (unsigned)B[(size_t)(k0 + r_b) * N + c0] : 0u;
        unsigned hi2 = (c1 < N) ? (unsigned)B[(size_t)(k0 + r_b) * N + c1] : 0u;
        bu[j] = lo | (hi2 << 16);
      }
    }
  };
  auto storeB = [&](int pb) {
    #pragma unroll
    for (int j = 0; j < 4; ++j) {
      BsT[pb][c_b + 2 * j][r_b]     = (unsigned short)(bu[j] & 0xffffu);
      BsT[pb][c_b + 2 * j + 1][r_b] = (unsigned short)(bu[j] >> 16);
    }
  };
#if GEMM_ASYNC
  auto issueA = [&](int tile) {
    #pragma unroll
    for (int half = 0; half < 2; ++half) {
      int r = half ? ra1 : ra0;
      int gr = m0 + r;
      void* lp = (void*)&As[tile & 3][r][ca];
      if (!MG || gr < M) {
        const unsigned short* gp = A + (size_t)gr * K + (tile << 5) + ca;
        __builtin_amdgcn_global_load_async_to_lds_b128((v4i*)gp, (v4i*)lp, 0, 0);
      } else {
        *(uint4*)lp = make_uint4(0u, 0u, 0u, 0u);
      }
    }
  };
#else
  auto loadA = [&](int k0) {
    #pragma unroll
    for (int half = 0; half < 2; ++half) {
      int r = half ? ra1 : ra0;
      int gr = m0 + r;
      au[half * 4 + 0] = au[half * 4 + 1] = au[half * 4 + 2] = au[half * 4 + 3] = 0u;
      if (!MG || gr < M) {
        const uint4* p = (const uint4*)(A + (size_t)gr * K + k0 + ca);
        uint4 t = *p;
        au[half * 4 + 0] = t.x; au[half * 4 + 1] = t.y;
        au[half * 4 + 2] = t.z; au[half * 4 + 3] = t.w;
      }
    }
  };
  auto storeA = [&](int pa) {
    *(uint4*)&As[pa][ra0][ca] = make_uint4(au[0], au[1], au[2], au[3]);
    *(uint4*)&As[pa][ra1][ca] = make_uint4(au[4], au[5], au[6], au[7]);
  };
#endif
  auto compute = [&](int pa, int pb) {
    Frag af0, af1, bf0, bf1;
    int mA0 = wr * 32 + l15;
    int mA1 = mA0 + 16;
    #pragma unroll
    for (int v = 0; v < 8; ++v) {
      int kk = ((v >> 2) << 4) + (hi << 3) + ((v & 3) << 1);
      af0.u[v] = *(const unsigned int*)&As[pa][mA0][kk];
      af1.u[v] = *(const unsigned int*)&As[pa][mA1][kk];
    }
    int c0 = wc * 32 + l15;
    #pragma unroll
    for (int v = 0; v < 8; ++v) {
      int kk = ((v >> 2) << 4) + (hi << 3) + ((v & 3) << 1);
      bf0.u[v] = *(const unsigned int*)&BsT[pb][c0][kk];
      bf1.u[v] = *(const unsigned int*)&BsT[pb][c0 + 16][kk];
    }
    acc[0][0] = wmma_bf16(af0, bf0, acc[0][0]);
    acc[0][1] = wmma_bf16(af0, bf1, acc[0][1]);
    acc[1][0] = wmma_bf16(af1, bf0, acc[1][0]);
    acc[1][1] = wmma_bf16(af1, bf1, acc[1][1]);
  };

  // ---- prologue ----
  loadB(0);
#if GEMM_ASYNC
  issueA(0);
  if (nt > 1) issueA(1);
#else
  loadA(0);
#endif

  // ---- steady state: branch-free pipeline ----
  for (int i = 0; i + 2 < nt; ++i) {
    storeB(i & 1);
#if !GEMM_ASYNC
    storeA(i & 3);
#endif
    loadB((i + 1) << 5);
#if GEMM_ASYNC
    issueA(i + 2);
    WAIT_ASYNC(4);            // tile i done; i+1, i+2 (2 issues each) in flight
#else
    loadA((i + 1) << 5);
#endif
    __syncthreads();
    compute(i & 3, i & 1);
  }
  // ---- tail: i = nt-2 ----
  if (nt >= 2) {
    int i = nt - 2;
    storeB(i & 1);
#if !GEMM_ASYNC
    storeA(i & 3);
#endif
    loadB((i + 1) << 5);
#if GEMM_ASYNC
    WAIT_ASYNC(2);
#else
    loadA((i + 1) << 5);
#endif
    __syncthreads();
    compute(i & 3, i & 1);
  }
  // ---- tail: i = nt-1 ----
  {
    int i = nt - 1;
    storeB(i & 1);
#if !GEMM_ASYNC
    storeA(i & 3);
#endif
#if GEMM_ASYNC
    WAIT_ASYNC(0);
#endif
    __syncthreads();
    compute(i & 3, i & 1);
  }

  // ---- epilogue ----
  #pragma unroll
  for (int tm = 0; tm < 2; ++tm) {
    #pragma unroll
    for (int v = 0; v < 8; ++v) {
      int row = m0 + wr * 32 + tm * 16 + v + hi * 8;
      if (MG && row >= M) continue;
      #pragma unroll
      for (int tn = 0; tn < 2; ++tn) {
        int colg = n0 + wc * 32 + tn * 16 + l15;
        if (NG && colg >= N) continue;
        float val = acc[tm][tn][v];
        if (rowbias) val += bf2f(rowbias[(size_t)(row >> rb_shift) * N + colg]);
        if (fbias)   val += fbias[colg];
        if (out_f32) ((float*)Cv)[(size_t)row * N + colg] = val;
        else ((unsigned short*)Cv)[(size_t)row * N + colg] = f2bf(val);
      }
    }
  }
}

// ---------------------------------------------------------------------------
// Attention: one workgroup per (b,h). Sq=64, head dim 64.
// scores fp32 in LDS -> softmax -> in-place bf16 P (front half of rows),
// V^T staged into the freed back half -> all fragments are dword LDS loads.
// Causal: fully-masked key tiles skip WMMA and write -inf directly.
// ---------------------------------------------------------------------------
template <int NKPAD, int CAUSAL>
__global__ __launch_bounds__(256) void k_attn(
    const unsigned short* __restrict__ Q,   // [B,64,512]
    const unsigned short* __restrict__ Kb,  // [B,NKrows,512]
    const unsigned short* __restrict__ Vb,  // [B,NKrows,512]
    unsigned short* __restrict__ O,         // [B,64,512]
    int H_, int NK, int NKrows) {
  __shared__ float sc[64][NKPAD];
  __shared__ float red[64][8];

  int bh = blockIdx.x;
  int b = bh / H_, h = bh % H_;
  const unsigned short* Qh = Q  + (size_t)b * 64 * 512 + h * 64;
  const unsigned short* Kh = Kb + (size_t)b * NKrows * 512 + h * 64;
  const unsigned short* Vh = Vb + (size_t)b * NKrows * 512 + h * 64;

  int tid = threadIdx.x, lane = tid & 31, w = tid >> 5;
  int wr = w & 3, wc = w >> 2, hi = lane >> 4, l15 = lane & 15;

  // hoisted Q fragments (invariant over key tiles)
  Frag aq[2];
  {
    int m = wr * 16 + l15;
    #pragma unroll
    for (int ks = 0; ks < 2; ++ks) {
      #pragma unroll
      for (int v = 0; v < 8; ++v) {
        int kk = ((v >> 2) << 4) + (hi << 3) + ((v & 3) << 1) + ks * 32;
        aq[ks].u[v] = *(const unsigned int*)(Qh + (size_t)m * 512 + kk);
      }
    }
  }

  // ---- scores = Q K^T / 8, masked ----
  constexpr int NT16 = NKPAD >> 4;
  for (int nt = wc; nt < NT16; nt += 2) {
    int key = nt * 16 + l15;
    if (CAUSAL && nt > wr) {           // fully-masked tile: no WMMA needed
      #pragma unroll
      for (int v = 0; v < 8; ++v) sc[wr * 16 + v + hi * 8][key] = -1e30f;
      continue;
    }
    v8f acc = {};
    #pragma unroll
    for (int ks = 0; ks < 2; ++ks) {
      Frag bf;
      #pragma unroll
      for (int v = 0; v < 8; ++v) {
        int kd = ((v >> 2) << 4) + (hi << 3) + ((v & 3) << 1) + ks * 32;
        bf.u[v] = (key < NK) ? *(const unsigned int*)(Kh + (size_t)key * 512 + kd) : 0u;
      }
      acc = wmma_bf16(aq[ks], bf, acc);
    }
    #pragma unroll
    for (int v = 0; v < 8; ++v) {
      int qrow = wr * 16 + v + hi * 8;
      float s = acc[v] * 0.125f;
      if (key >= NK || (CAUSAL && key > qrow)) s = -1e30f;
      sc[qrow][key] = s;
    }
  }
  __syncthreads();

  // ---- row softmax (4 threads/row, register-buffered) -> bf16 P in place ----
  {
    constexpr int CH = NKPAD / 4;
    int row = tid >> 2, j = tid & 3;
    float loc[CH];
    float mx = -1e30f;
    #pragma unroll
    for (int i = 0; i < CH; ++i) { loc[i] = sc[row][j * CH + i]; mx = fmaxf(mx, loc[i]); }
    red[row][j] = mx;
    __syncthreads();
    mx = fmaxf(fmaxf(red[row][0], red[row][1]), fmaxf(red[row][2], red[row][3]));
    float sum = 0.f;
    #pragma unroll
    for (int i = 0; i < CH; ++i) { float e = __expf(loc[i] - mx); loc[i] = e; sum += e; }
    red[row][4 + j] = sum;
    __syncthreads();
    sum = red[row][4] + red[row][5] + red[row][6] + red[row][7];
    float inv = 1.0f / sum;
    unsigned short* prow = (unsigned short*)&sc[row][0];
    #pragma unroll
    for (int i = 0; i < CH; ++i) prow[j * CH + i] = f2bf(loc[i] * inv);
  }
  __syncthreads();

  // ---- stage V^T into the freed back half of each sc row ----
  unsigned short* T = (unsigned short*)&sc[0][0];
  for (int vv = tid; vv < NKPAD * 8; vv += 256) {
    int key = vv >> 3;
    int c8  = (vv & 7) * 8;
    unsigned bu0 = 0, bu1 = 0, bu2 = 0, bu3 = 0;
    if (key < NK) {
      const uint4* p = (const uint4*)(Vh + (size_t)key * 512 + c8);
      uint4 t = *p;
      bu0 = t.x; bu1 = t.y; bu2 = t.z; bu3 = t.w;
    }
    unsigned buv[4] = {bu0, bu1, bu2, bu3};
    #pragma unroll
    for (int i = 0; i < 4; ++i) {
      T[(size_t)(c8 + 2 * i)     * (2 * NKPAD) + NKPAD + key] = (unsigned short)(buv[i] & 0xffffu);
      T[(size_t)(c8 + 2 * i + 1) * (2 * NKPAD) + NKPAD + key] = (unsigned short)(buv[i] >> 16);
    }
  }
  __syncthreads();

  // ---- O = P V : all-LDS dword fragments ----
  constexpr int NT32 = NKPAD >> 5;
  constexpr int PSTRIDE = NKPAD * 2;
  v8f o0 = {}, o1 = {};
  for (int kb = 0; kb < NT32; ++kb) {
    Frag pf;
    int m = wr * 16 + l15;
    #pragma unroll
    for (int v = 0; v < 8; ++v) {
      int kk = kb * 32 + ((v >> 2) << 4) + (hi << 3) + ((v & 3) << 1);
      pf.u[v] = *(const unsigned int*)&T[(size_t)m * PSTRIDE + kk];
    }
    #pragma unroll
    for (int t = 0; t < 2; ++t) {
      Frag vf;
      int dv = wc * 32 + t * 16 + l15;
      const unsigned short* Trow = T + (size_t)dv * PSTRIDE + NKPAD;
      #pragma unroll
      for (int v = 0; v < 8; ++v) {
        int key = kb * 32 + ((v >> 2) << 4) + (hi << 3) + ((v & 3) << 1);
        vf.u[v] = *(const unsigned int*)&Trow[key];
      }
      if (t == 0) o0 = wmma_bf16(pf, vf, o0);
      else        o1 = wmma_bf16(pf, vf, o1);
    }
  }
  #pragma unroll
  for (int v = 0; v < 8; ++v) {
    int qrow = wr * 16 + v + hi * 8;
    O[((size_t)b * 64 + qrow) * 512 + h * 64 + wc * 32 + l15]      = f2bf(o0[v]);
    O[((size_t)b * 64 + qrow) * 512 + h * 64 + wc * 32 + 16 + l15] = f2bf(o1[v]);
  }
}

// ---------------------------------------------------------------------------
// In-place row softmax (final vocab softmax), one block per row
// ---------------------------------------------------------------------------
__global__ __launch_bounds__(256) void k_softmax_rows(float* __restrict__ X,
                                                      int ncols) {
  __shared__ float red[256];
  float* row = X + (size_t)blockIdx.x * ncols;
  int tid = threadIdx.x;
  float mx = -1e30f;
  for (int i = tid; i < ncols; i += 256) mx = fmaxf(mx, row[i]);
  red[tid] = mx; __syncthreads();
  for (int s = 128; s > 0; s >>= 1) {
    if (tid < s) red[tid] = fmaxf(red[tid], red[tid + s]);
    __syncthreads();
  }
  mx = red[0]; __syncthreads();
  float sum = 0.f;
  for (int i = tid; i < ncols; i += 256) { float e = __expf(row[i] - mx); row[i] = e; sum += e; }
  red[tid] = sum; __syncthreads();
  for (int s = 128; s > 0; s >>= 1) {
    if (tid < s) red[tid] += red[tid + s];
    __syncthreads();
  }
  float inv = 1.0f / red[0];
  for (int i = tid; i < ncols; i += 256) row[i] *= inv;
}

// ---------------------------------------------------------------------------
// Host orchestration
// ---------------------------------------------------------------------------
extern "C" void kernel_launch(void* const* d_in, const int* in_sizes, int n_in,
                              void* d_out, int out_size, void* d_ws, size_t ws_size,
                              hipStream_t stream) {
  (void)in_sizes; (void)n_in; (void)out_size; (void)ws_size;
  constexpr int Bq = 32, S = 64, D = 512, H = 8, NR = 196, L = 6, VOCAB = 10000;
  constexpr int MS = Bq * S;
  constexpr int MR = Bq * NR;
  constexpr size_t WSZ = (size_t)D * D;

  const int*   x    = (const int*)  d_in[0];
  const float* Kf   = (const float*)d_in[1];
  const float* Vf   = (const float*)d_in[2];
  const float* gf   = (const float*)d_in[3];
  const float* emb  = (const float*)d_in[5];
  const float* Wq_s = (const float*)d_in[6];
  const float* Wk_s = (const float*)d_in[7];
  const float* Wv_s = (const float*)d_in[8];
  const float* Wo_s = (const float*)d_in[9];
  const float* Wq_c = (const float*)d_in[10];
  const float* Wg_c = (const float*)d_in[11];
  const float* Wk_c = (const float*)d_in[12];
  const float* Wv_c = (const float*)d_in[13];
  const float* Wo_c = (const float*)d_in[14];
  const float* Wf   = (const float*)d_in[15];
  const float* bf_  = (const float*)d_in[16];

  char* wsp = (char*)d_ws;
  auto alloc = [&](size_t bytes) -> void* {
    void* p = (void*)wsp;
    wsp += (bytes + 255) & ~(size_t)255;
    return p;
  };
  typedef unsigned short u16;
  u16* wq_s = (u16*)alloc(L * WSZ * 2);
  u16* wk_s = (u16*)alloc(L * WSZ * 2);
  u16* wv_s = (u16*)alloc(L * WSZ * 2);
  u16* wo_s = (u16*)alloc(L * WSZ * 2);
  u16* wq_c = (u16*)alloc(L * WSZ * 2);
  u16* wg_c = (u16*)alloc(L * WSZ * 2);
  u16* wk_c = (u16*)alloc(L * WSZ * 2);
  u16* wv_c = (u16*)alloc(L * WSZ * 2);
  u16* wo_c = (u16*)alloc(L * WSZ * 2);
  u16* wfb  = (u16*)alloc((size_t)D * VOCAB * 2);
  u16* kin  = (u16*)alloc((size_t)MR * D * 2);
  u16* vin  = (u16*)alloc((size_t)MR * D * 2);
  u16* gin  = (u16*)alloc((size_t)Bq * D * 2);
  u16* X    = (u16*)alloc((size_t)MS * D * 2);
  u16* Qb   = (u16*)alloc((size_t)MS * D * 2);
  u16* Kb2  = (u16*)alloc((size_t)MS * D * 2);
  u16* Vb2  = (u16*)alloc((size_t)MS * D * 2);
  u16* AttO = (u16*)alloc((size_t)MS * D * 2);
  u16* X2   = (u16*)alloc((size_t)MS * D * 2);
  u16* Gq   = (u16*)alloc((size_t)Bq * D * 2);
  u16* Qc   = (u16*)alloc((size_t)MS * D * 2);
  u16* Kc   = (u16*)alloc((size_t)MR * D * 2);
  u16* Vc   = (u16*)alloc((size_t)MR * D * 2);
  u16* CrO  = (u16*)alloc((size_t)MS * D * 2);

  auto cvt = [&](const float* in, u16* out, int n) {
    int blocks = (n + 256 * 8 - 1) / (256 * 8);
    k_f32_to_bf16<<<blocks, 256, 0, stream>>>(in, out, n);
  };
  auto gemm = [&](const u16* A, const u16* Bm, void* C, int M, int N, int K,
                  const u16* rowbias, int rb_shift, const float* fbias, int out_f32) {
    dim3 grid((N + 63) / 64, (M + 127) / 128);
    bool mfull = (M % 128) == 0;
    bool nfull = (N % 64) == 0;
    if (mfull && nfull)
      k_gemm<0, 0><<<grid, 256, 0, stream>>>(A, Bm, C, M, N, K, rowbias, rb_shift, fbias, out_f32);
    else if (!mfull && nfull)
      k_gemm<1, 0><<<grid, 256, 0, stream>>>(A, Bm, C, M, N, K, rowbias, rb_shift, fbias, out_f32);
    else if (mfull && !nfull)
      k_gemm<0, 1><<<grid, 256, 0, stream>>>(A, Bm, C, M, N, K, rowbias, rb_shift, fbias, out_f32);
    else
      k_gemm<1, 1><<<grid, 256, 0, stream>>>(A, Bm, C, M, N, K, rowbias, rb_shift, fbias, out_f32);
  };

  cvt(Wq_s, wq_s, L * WSZ); cvt(Wk_s, wk_s, L * WSZ); cvt(Wv_s, wv_s, L * WSZ);
  cvt(Wo_s, wo_s, L * WSZ); cvt(Wq_c, wq_c, L * WSZ); cvt(Wg_c, wg_c, L * WSZ);
  cvt(Wk_c, wk_c, L * WSZ); cvt(Wv_c, wv_c, L * WSZ); cvt(Wo_c, wo_c, L * WSZ);
  cvt(Wf, wfb, D * VOCAB);
  cvt(Kf, kin, MR * D); cvt(Vf, vin, MR * D); cvt(gf, gin, Bq * D);

  k_embed<<<MS, 256, 0, stream>>>(x, emb, X, S, D);

  for (int l = 0; l < L; ++l) {
    const u16* wq = wq_s + (size_t)l * WSZ; const u16* wk = wk_s + (size_t)l * WSZ;
    const u16* wv = wv_s + (size_t)l * WSZ; const u16* wo = wo_s + (size_t)l * WSZ;
    const u16* cq = wq_c + (size_t)l * WSZ; const u16* cg = wg_c + (size_t)l * WSZ;
    const u16* ck = wk_c + (size_t)l * WSZ; const u16* cv = wv_c + (size_t)l * WSZ;
    const u16* co = wo_c + (size_t)l * WSZ;

    gemm(X, wq, Qb,  MS, D, D, nullptr, 0, nullptr, 0);
    gemm(X, wk, Kb2, MS, D, D, nullptr, 0, nullptr, 0);
    gemm(X, wv, Vb2, MS, D, D, nullptr, 0, nullptr, 0);
    k_attn<64, 1><<<Bq * H, 256, 0, stream>>>(Qb, Kb2, Vb2, AttO, H, S, S);
    gemm(AttO, wo, X2, MS, D, D, nullptr, 0, nullptr, 0);

    gemm(gin, cg, Gq, Bq, D, D, nullptr, 0, nullptr, 0);
    gemm(X2, cq, Qc, MS, D, D, Gq, 6, nullptr, 0);
    gemm(kin, ck, Kc, MR, D, D, nullptr, 0, nullptr, 0);
    gemm(vin, cv, Vc, MR, D, D, nullptr, 0, nullptr, 0);
    k_attn<224, 0><<<Bq * H, 256, 0, stream>>>(Qc, Kc, Vc, CrO, H, NR, NR);
    gemm(CrO, co, X, MS, D, D, nullptr, 0, nullptr, 0);
  }

  gemm(X, wfb, d_out, MS, VOCAB, D, nullptr, 0, bf_, 1);
  k_softmax_rows<<<MS, 256, 0, stream>>>((float*)d_out, VOCAB);
}
